// oldGCN_64424509440201
// MI455X (gfx1250) — matrix operand order repaired and checked by reading, not compiled, not run on hardware
//
#include <hip/hip_runtime.h>
#include <hip/hip_bf16.h>

// CDNA5 (gfx1250) wave32 WMMA types
typedef __attribute__((ext_vector_type(16))) _Float16 v16h;
typedef __attribute__((ext_vector_type(8)))  float    v8f;

#define HDIM 64
#define DOUTC 16

#define WMMA_F16(a, b, c) \
    __builtin_amdgcn_wmma_f32_16x16x32_f16(false, (a), false, (b), (short)0, (c), false, false)

// ---------------------------------------------------------------------------
// Pack a row-major f32 weight [K x N] into the f16 B-fragment layout consumed
// by v_wmma_f32_16x16x32_f16. Per (k-chunk of 32, n-tile of 16): 32 lanes x
// v16h.  B layout (16-bit, 32x16 KxN): col = lane%16, element i holds
// K = chunk*32 + (i<8 ? (lane/16)*8 + i : 16 + (lane/16)*8 + (i-8)).
// ---------------------------------------------------------------------------
__global__ void pack_w_kernel(const float* __restrict__ W,
                              _Float16* __restrict__ Wp, int K, int N) {
    int idx = blockIdx.x * blockDim.x + threadIdx.x;
    if (idx >= K * N) return;
    int i    = idx & 15;          // element within v16h
    int lane = (idx >> 4) & 31;   // lane
    int ct   = idx >> 9;          // chunk*NT + t
    int NT   = N >> 4;
    int t     = ct % NT;
    int chunk = ct / NT;
    int r    = lane & 15;
    int half = lane >> 4;
    int k = chunk * 32 + (i < 8 ? half * 8 + i : 16 + half * 8 + (i - 8));
    int n = t * 16 + r;
    Wp[idx] = (_Float16)W[(size_t)k * N + n];
}

// ---------------------------------------------------------------------------
// Degree / symmetric normalization: deg[i] = 1 (self loop) + indegree(i),
// then dinv[i] = rsqrt(deg[i])  (in place).
// ---------------------------------------------------------------------------
__global__ void deg_init_kernel(float* __restrict__ deg, int n) {
    int i = blockIdx.x * blockDim.x + threadIdx.x;
    if (i < n) deg[i] = 1.0f;
}
__global__ void deg_count_kernel(const int* __restrict__ dst,
                                 float* __restrict__ deg, int e) {
    int i = blockIdx.x * blockDim.x + threadIdx.x;
    if (i < e) atomicAdd(deg + dst[i], 1.0f);
}
__global__ void dinv_kernel(float* __restrict__ deg, int n) {
    int i = blockIdx.x * blockDim.x + threadIdx.x;
    if (i < n) {
        float d = deg[i];
        deg[i] = (d > 0.0f) ? rsqrtf(d) : 0.0f;
    }
}

// ---------------------------------------------------------------------------
// GEMM (M x K) @ (K x 64), WMMA f16->f32.  Each wave owns a 32x64 block (two
// 16-row tiles sharing the 4 B fragments -> 8 WMMAs per 32-deep K step).
// Requires nNodes % 16 == 0 (50000 = 3125*16): no bounds guards, so the
// epilogue is straight-line and EXEC stays all-ones throughout.
// Epilogue scales rows by dinv and writes hs (gather source) and acc
// (accumulator pre-loaded with the self-loop term).
// ---------------------------------------------------------------------------
__global__ __launch_bounds__(128) void gemm_hs_kernel(
    const float* __restrict__ A, const _Float16* __restrict__ Wp,
    const float* __restrict__ dinv, float* __restrict__ hs,
    float* __restrict__ acc, int K, int nNodes) {
    const int wave   = threadIdx.x >> 5;
    const int lane   = threadIdx.x & 31;
    const int mtiles = nNodes >> 4;
    const int mt0    = (blockIdx.x * 4 + wave) * 2;
    if (mt0 >= mtiles) return;                 // wave-uniform
    const bool has1  = (mt0 + 1) < mtiles;
    const int mBase0 = mt0 << 4;
    const int mBase1 = mBase0 + 16;
    const int r      = lane & 15;
    const int half   = lane >> 4;

    const float* __restrict__ xrow0 = A + (size_t)(mBase0 + r) * K;
    const float* __restrict__ xrow1 = A + (size_t)((has1 ? mBase1 : mBase0) + r) * K;

    v8f c00 = {}, c01 = {}, c02 = {}, c03 = {};
    v8f c10 = {}, c11 = {}, c12 = {}, c13 = {};
    const int chunks = K >> 5;
    for (int ch = 0; ch < chunks; ++ch) {
        const int kb = ch << 5;
        v16h a0, a1;
#pragma unroll
        for (int v = 0; v < 8; ++v) {
            // 16-bit A 16x32 layout: v<4 -> K = 2v + 8*half ; v>=4 -> 16+2(v-4)+8*half
            const int k = kb + (v < 4 ? (v * 2 + half * 8)
                                      : (16 + (v - 4) * 2 + half * 8));
            const float2 p0 = *(const float2*)(xrow0 + k);
            const float2 p1 = *(const float2*)(xrow1 + k);
            a0[2 * v]     = (_Float16)p0.x;
            a0[2 * v + 1] = (_Float16)p0.y;
            a1[2 * v]     = (_Float16)p1.x;
            a1[2 * v + 1] = (_Float16)p1.y;
        }
        const _Float16* bp = Wp + ((size_t)(ch * 4) * 32 + lane) * 16;
        const v16h b0 = *(const v16h*)(bp);
        const v16h b1 = *(const v16h*)(bp + 512);
        const v16h b2 = *(const v16h*)(bp + 1024);
        const v16h b3 = *(const v16h*)(bp + 1536);
        c00 = WMMA_F16(a0, b0, c00);
        c10 = WMMA_F16(a1, b0, c10);
        c01 = WMMA_F16(a0, b1, c01);
        c11 = WMMA_F16(a1, b1, c11);
        c02 = WMMA_F16(a0, b2, c02);
        c12 = WMMA_F16(a1, b2, c12);
        c03 = WMMA_F16(a0, b3, c03);
        c13 = WMMA_F16(a1, b3, c13);
    }

    // C/D layout: element i of lane {r,half}: row = mBase + 8*half + i, col = r.
    // Rows are contiguous -> dinv comes in as two float4 loads.
    auto store_block = [&](int mBase, const v8f& t0, const v8f& t1,
                           const v8f& t2, const v8f& t3) {
        const int rb = mBase + 8 * half;
        const float4 dva = *(const float4*)(dinv + rb);
        const float4 dvb = *(const float4*)(dinv + rb + 4);
        const float dv[8] = { dva.x, dva.y, dva.z, dva.w,
                              dvb.x, dvb.y, dvb.z, dvb.w };
#pragma unroll
        for (int i = 0; i < 8; ++i) {
            const size_t base = (size_t)(rb + i) * HDIM + r;
            float v0 = t0[i] * dv[i];
            float v1 = t1[i] * dv[i];
            float v2 = t2[i] * dv[i];
            float v3 = t3[i] * dv[i];
            hs[base]       = v0;  acc[base]       = v0;
            hs[base + 16]  = v1;  acc[base + 16]  = v1;
            hs[base + 32]  = v2;  acc[base + 32]  = v2;
            hs[base + 48]  = v3;  acc[base + 48]  = v3;
        }
    };
    store_block(mBase0, c00, c01, c02, c03);
    if (has1) store_block(mBase1, c10, c11, c12, c13);
}

// ---------------------------------------------------------------------------
// Head GEMM: (M x 64) @ (64 x 16) -> logits + bias.
// ---------------------------------------------------------------------------
__global__ __launch_bounds__(128) void gemm_logits_kernel(
    const float* __restrict__ A, const _Float16* __restrict__ Wp,
    const float* __restrict__ bl, float* __restrict__ out, int nNodes) {
    const int wave   = threadIdx.x >> 5;
    const int lane   = threadIdx.x & 31;
    const int mtiles = nNodes >> 4;
    const int mt0    = (blockIdx.x * 4 + wave) * 2;
    if (mt0 >= mtiles) return;
    const bool has1  = (mt0 + 1) < mtiles;
    const int mBase0 = mt0 << 4;
    const int mBase1 = mBase0 + 16;
    const int r      = lane & 15;
    const int half   = lane >> 4;

    const float* __restrict__ xrow0 = A + (size_t)(mBase0 + r) * HDIM;
    const float* __restrict__ xrow1 = A + (size_t)((has1 ? mBase1 : mBase0) + r) * HDIM;

    v8f c0 = {}, c1 = {};
#pragma unroll
    for (int ch = 0; ch < 2; ++ch) {
        const int kb = ch << 5;
        v16h a0, a1;
#pragma unroll
        for (int v = 0; v < 8; ++v) {
            const int k = kb + (v < 4 ? (v * 2 + half * 8)
                                      : (16 + (v - 4) * 2 + half * 8));
            const float2 p0 = *(const float2*)(xrow0 + k);
            const float2 p1 = *(const float2*)(xrow1 + k);
            a0[2 * v]     = (_Float16)p0.x;
            a0[2 * v + 1] = (_Float16)p0.y;
            a1[2 * v]     = (_Float16)p1.x;
            a1[2 * v + 1] = (_Float16)p1.y;
        }
        const v16h b = *(const v16h*)(Wp + ((size_t)ch * 32 + lane) * 16);
        c0 = WMMA_F16(a0, b, c0);
        c1 = WMMA_F16(a1, b, c1);
    }
    const float bias = bl[r];
#pragma unroll
    for (int i = 0; i < 8; ++i) {
        out[(size_t)(mBase0 + 8 * half + i) * DOUTC + r] = c0[i] + bias;
    }
    if (has1) {
#pragma unroll
        for (int i = 0; i < 8; ++i) {
            out[(size_t)(mBase1 + 8 * half + i) * DOUTC + r] = c1[i] + bias;
        }
    }
}

// ---------------------------------------------------------------------------
// Edge scatter: one wave per edge; each lane carries 2 of the 64 features.
// Coalesced 256B gather of hs[src], f32 atomics into acc[dst] (lands in L2).
// ---------------------------------------------------------------------------
__global__ __launch_bounds__(256) void edge_scatter_kernel(
    const int* __restrict__ src, const int* __restrict__ dst,
    const float* __restrict__ hs, float* __restrict__ acc, int E) {
    int t = blockIdx.x * blockDim.x + threadIdx.x;
    int e = t >> 5;
    if (e >= E) return;
    int lane = t & 31;
    int s = src[e];
    int d = dst[e];
    const float2 v = *(const float2*)(hs + (size_t)s * HDIM + lane * 2);
    float* p = acc + (size_t)d * HDIM + lane * 2;
    atomicAdd(p,     v.x);
    atomicAdd(p + 1, v.y);
}

// ---------------------------------------------------------------------------
// Fused normalize + bias + ReLU:  a = relu(dinv[row]*acc + b[f])  (float4)
// ---------------------------------------------------------------------------
__global__ __launch_bounds__(256) void epilogue_relu_kernel(
    const float* __restrict__ acc, const float* __restrict__ dinv,
    const float* __restrict__ bias, float* __restrict__ out, int total) {
    int idx = (blockIdx.x * blockDim.x + threadIdx.x) * 4;
    if (idx >= total) return;
    float4 v = *(const float4*)(acc + idx);
    int row = idx >> 6;    // /64
    int f   = idx & 63;
    float s = dinv[row];
    float4 bb = *(const float4*)(bias + f);
    v.x = fmaxf(fmaf(v.x, s, bb.x), 0.0f);
    v.y = fmaxf(fmaf(v.y, s, bb.y), 0.0f);
    v.z = fmaxf(fmaf(v.z, s, bb.z), 0.0f);
    v.w = fmaxf(fmaf(v.w, s, bb.w), 0.0f);
    *(float4*)(out + idx) = v;
}

// ---------------------------------------------------------------------------
// Row softmax over 16 classes: one thread per row.
// ---------------------------------------------------------------------------
__global__ __launch_bounds__(256) void softmax_kernel(
    const float* __restrict__ logits, float* __restrict__ probs, int n) {
    int row = blockIdx.x * blockDim.x + threadIdx.x;
    if (row >= n) return;
    const float* l = logits + (size_t)row * DOUTC;
    float m = l[0];
#pragma unroll
    for (int i = 1; i < DOUTC; ++i) m = fmaxf(m, l[i]);
    float e[DOUTC];
    float s = 0.0f;
#pragma unroll
    for (int i = 0; i < DOUTC; ++i) { e[i] = expf(l[i] - m); s += e[i]; }
    float inv = 1.0f / s;
    float* p = probs + (size_t)row * DOUTC;
#pragma unroll
    for (int i = 0; i < DOUTC; ++i) p[i] = e[i] * inv;
}

// ---------------------------------------------------------------------------
extern "C" void kernel_launch(void* const* d_in, const int* in_sizes, int n_in,
                              void* d_out, int out_size, void* d_ws, size_t ws_size,
                              hipStream_t stream) {
    const float* x  = (const float*)d_in[0];
    const int*   ei = (const int*)d_in[1];
    const float* W1 = (const float*)d_in[2];
    const float* b1 = (const float*)d_in[3];
    const float* W2 = (const float*)d_in[4];
    const float* b2 = (const float*)d_in[5];
    const float* Wl = (const float*)d_in[6];
    const float* bl = (const float*)d_in[7];
    float* out = (float*)d_out;

    const int DIN = 256;
    const int nNodes = in_sizes[0] / DIN;   // 50000, multiple of 16
    const int E = in_sizes[1] / 2;
    const int* src = ei;
    const int* dst = ei + E;

    // Workspace carve-out (256B aligned)
    char* ws = (char*)d_ws;
    size_t off = 0;
    auto carve = [&](size_t bytes) -> void* {
        void* p = ws + off;
        off += (bytes + 255) & ~(size_t)255;
        return p;
    };
    float*    dinv = (float*)carve((size_t)nNodes * 4);
    _Float16* W1p  = (_Float16*)carve((size_t)DIN * HDIM * 2);
    _Float16* W2p  = (_Float16*)carve((size_t)HDIM * HDIM * 2);
    _Float16* Wlp  = (_Float16*)carve((size_t)HDIM * DOUTC * 2);
    float*    hs   = (float*)carve((size_t)nNodes * HDIM * 4);
    float*    acc  = (float*)carve((size_t)nNodes * HDIM * 4);
    float*    act  = (float*)carve((size_t)nNodes * HDIM * 4);

    // Weight packing into WMMA B-fragment layout (cheap, once per call)
    pack_w_kernel<<<(DIN * HDIM + 255) / 256, 256, 0, stream>>>(W1, W1p, DIN, HDIM);
    pack_w_kernel<<<(HDIM * HDIM + 255) / 256, 256, 0, stream>>>(W2, W2p, HDIM, HDIM);
    pack_w_kernel<<<(HDIM * DOUTC + 255) / 256, 256, 0, stream>>>(Wl, Wlp, HDIM, DOUTC);

    // Symmetric normalization coefficients
    deg_init_kernel<<<(nNodes + 255) / 256, 256, 0, stream>>>(dinv, nNodes);
    deg_count_kernel<<<(E + 255) / 256, 256, 0, stream>>>(dst, dinv, E);
    dinv_kernel<<<(nNodes + 255) / 256, 256, 0, stream>>>(dinv, nNodes);

    const int mtiles   = nNodes / 16;           // 3125
    const int pairs    = (mtiles + 1) / 2;      // 32 rows per wave
    const int gemmGrid = (pairs + 3) / 4;       // 4 waves per block
    const int edgeGrid = (int)(((size_t)E * 32 + 255) / 256);
    const int epiGrid  = (int)(((size_t)nNodes * HDIM / 4 + 255) / 256);

    // Layer 1: hs = (x@W1)*dinv ; acc = hs ; acc += gather ; act = relu(dinv*acc+b1)
    gemm_hs_kernel<<<gemmGrid, 128, 0, stream>>>(x, W1p, dinv, hs, acc, DIN, nNodes);
    edge_scatter_kernel<<<edgeGrid, 256, 0, stream>>>(src, dst, hs, acc, E);
    epilogue_relu_kernel<<<epiGrid, 256, 0, stream>>>(acc, dinv, b1, act, nNodes * HDIM);

    // Layer 2
    gemm_hs_kernel<<<gemmGrid, 128, 0, stream>>>(act, W2p, dinv, hs, acc, HDIM, nNodes);
    edge_scatter_kernel<<<edgeGrid, 256, 0, stream>>>(src, dst, hs, acc, E);
    epilogue_relu_kernel<<<epiGrid, 256, 0, stream>>>(acc, dinv, b2, act, nNodes * HDIM);

    // Head + softmax
    gemm_logits_kernel<<<gemmGrid, 128, 0, stream>>>(act, Wlp, bl, out, nNodes);
    softmax_kernel<<<(nNodes + 255) / 256, 256, 0, stream>>>(out, out + (size_t)nNodes * DOUTC, nNodes);
}
